// SetPredictor_74371653697803
// MI455X (gfx1250) — matrix operand order repaired and checked by reading, not compiled, not run on hardware
//
#include <hip/hip_runtime.h>
#include <math.h>

// ---------------------------------------------------------------------------
// DiT-style transformer for MI455X (gfx1250), bf16 WMMA everywhere.
// B=8 T=32 N=32 D=512 DEPTH=4 H=8 DH=64 MLP=2048, L=1024, rows = B*L = 8192.
// A-tile GEMM staging uses the Tensor Data Mover (tensor_load_to_lds) with
// double buffering on TENSORcnt when the builtin is available.
// ---------------------------------------------------------------------------

typedef unsigned int   u32;
typedef unsigned short u16b;
typedef __attribute__((ext_vector_type(16))) __bf16 v16bf;
typedef __attribute__((ext_vector_type(8)))  float  v8f;
typedef __attribute__((ext_vector_type(4)))  unsigned int v4u;
typedef __attribute__((ext_vector_type(8)))  int    v8i;
typedef __attribute__((ext_vector_type(4)))  int    v4i;

#if defined(__has_builtin)
#if __has_builtin(__builtin_amdgcn_tensor_load_to_lds) && \
    __has_builtin(__builtin_amdgcn_s_wait_tensorcnt)
#define USE_TDM 1
#endif
#endif
#ifndef USE_TDM
#define USE_TDM 0
#endif

union FragB { u32 u[8]; v16bf v; };

__device__ __forceinline__ v8f zero_v8f() {
  v8f z = {0.f, 0.f, 0.f, 0.f, 0.f, 0.f, 0.f, 0.f};
  return z;
}

__device__ __forceinline__ u16b f2bf(float f) {
  union { float f; u32 u; } c; c.f = f;
  u32 u = c.u;
  return (u16b)((u + 0x7FFFu + ((u >> 16) & 1u)) >> 16);  // round-to-nearest-even
}

#if USE_TDM
// TDM: load a 64-row x 32-elem bf16 tile of A (row stride K elems) into LDS.
// LDS layout gets 1 DWORD of padding after every 16 DWORDs (pad_interval=3,
// pad_amount=0 -> 1 DW), producing the stride-17-DWORD conflict-free layout
// that the WMMA fragment loaders below expect.  D# packing per ISA 8.3/8.4.
// This toolchain (clang-23 / therock-10.0) exposes the 6-arg builtin:
// (uint32x4 g0, int32x8 g1, int32x4 g2, int32x4 g3, int32x8, i32 cpol).
__device__ __forceinline__ void tdm_load_a_tile(const u16b* A, int bm, int k0,
                                                int M, int K, u32 lds_off) {
  unsigned long long ga =
      (unsigned long long)(size_t)(A + (size_t)bm * K + k0);
  v4u g0 = {1u,                                   // count=1 (valid user D#)
            lds_off,                              // lds_addr (bytes)
            (u32)ga,                              // global_addr[31:0]
            (u32)(ga >> 32) | (2u << 30)};        // global_addr[56:32] | type=2
  v8i g1;
  g1[0] = (int)((1u << 16)      // data_size = 1 -> 2 bytes
                | (1u << 20)    // pad_enable
                | (3u << 22));  // pad_interval = 3 -> every 16 DWORDs
                                // pad_amount = 0 -> 1 DWORD
  g1[1] = (int)(((u32)K & 0xFFFFu) << 16);                    // tensor_dim0 lo16
  g1[2] = (int)(((u32)K >> 16) | (((u32)M & 0xFFFFu) << 16)); // dim0 hi / dim1 lo
  g1[3] = (int)(((u32)M >> 16) | (32u << 16));                // dim1 hi | tile_dim0=32
  g1[4] = 64;                                                 // tile_dim1=64, tile_dim2=0
  g1[5] = K;                                                  // tensor_dim0_stride lo32
  g1[6] = 0;                                                  // stride hi | dim1_stride lo
  g1[7] = 0;
  v4i z4 = {0, 0, 0, 0};
  v8i z8 = {0, 0, 0, 0, 0, 0, 0, 0};
  __builtin_amdgcn_tensor_load_to_lds(g0, g1, z4, z4, z8, 0);
}
#endif

// ---------------------------------------------------------------------------
// Generic bf16 GEMM: C(M,N) = A(M,K) @ B(K,N), A/B row-major bf16 (ushort).
// 64x64 block tile, BK=32, 128 threads = 4 waves, each wave 32x32 (2x2 WMMA).
// A tile staged by TDM (double-buffered on TENSORcnt) when available.
// MODE 0: outF[m,n] = acc + bias[n]                          (adaLN mod)
// MODE 1: outB[m,n] = bf16(acc)                              (QKV)
// MODE 2: outF[m,n] += mod[(m>>5)*3072+gate_off+n]*(acc+bias[n])  (residual)
// MODE 3: outB[m,n] = bf16(gelu_exact(acc + bias[n]))        (MLP1)
// ---------------------------------------------------------------------------
template <int MODE>
__global__ __launch_bounds__(128) void gemm_bf16(
    const u16b* __restrict__ A, const u16b* __restrict__ B,
    int M, int N, int K,
    const float* __restrict__ bias, const float* __restrict__ mod, int gate_off,
    float* __restrict__ outF, u16b* __restrict__ outB) {
  __shared__ u32 As32[2][64 * 17];  // [buf][row][kpair], stride 17 conflict-free
  __shared__ u32 Bs32[64 * 17];     // [col n][kpair], B transposed into LDS
  u16b* Bsh = (u16b*)Bs32;

  const int tid   = threadIdx.x;
  const int lane  = tid & 31;
  const int w     = tid >> 5;
  const int hi    = lane >> 4;    // half-wave: 0 or 1
  const int l15   = lane & 15;
  const int bm    = blockIdx.x * 64;
  const int bn    = blockIdx.y * 64;
  const int warpM = (w & 1) * 32;
  const int warpN = (w >> 1) * 32;

  v8f acc[2][2];
#pragma unroll
  for (int i = 0; i < 2; ++i)
#pragma unroll
    for (int j = 0; j < 2; ++j) acc[i][j] = zero_v8f();

#if USE_TDM
  if (w == 0) tdm_load_a_tile(A, bm, 0, M, K, (u32)(size_t)&As32[0][0]);
#endif

  for (int k0 = 0; k0 < K; k0 += 32) {
    const int cur = (k0 >> 5) & 1;
    __syncthreads();  // previous iteration's LDS readers are done
    // ---- stage B tile (32x64), transposed to [n][k] ----
#pragma unroll
    for (int c = tid; c < 256; c += 128) {
      int krow = c >> 3;
      int n8   = (c & 7) * 8;
      uint4 d = *(const uint4*)(B + (size_t)(k0 + krow) * N + bn + n8);
      u16b tmp[8];
      *(uint4*)tmp = d;
#pragma unroll
      for (int j = 0; j < 8; ++j) Bsh[(n8 + j) * 34 + krow] = tmp[j];
    }
    if (k0 + 32 < K)  // gfx1250 global_prefetch_b8 for the next B tile
      __builtin_prefetch(B + (size_t)(k0 + 32 + (tid >> 2)) * N + bn, 0, 1);
#if USE_TDM
    if (w == 0) {  // uniform per wave; TDM ignores EXEC -> one DMA per WG
      if (k0 + 32 < K) {
        tdm_load_a_tile(A, bm, k0 + 32, M, K, (u32)(size_t)&As32[cur ^ 1][0]);
        __builtin_amdgcn_s_wait_tensorcnt(1);  // in-order: current tile done
      } else {
        __builtin_amdgcn_s_wait_tensorcnt(0);
      }
    }
#else
    // ---- manual A tile staging (64x32) ----
#pragma unroll
    for (int c = tid; c < 256; c += 128) {
      int row  = c >> 2;
      int col8 = (c & 3) * 8;
      uint4 d = *(const uint4*)(A + (size_t)(bm + row) * K + k0 + col8);
      u32* dst = &As32[cur][row * 17 + (col8 >> 1)];
      dst[0] = d.x; dst[1] = d.y; dst[2] = d.z; dst[3] = d.w;
    }
    if (k0 + 32 < K)
      __builtin_prefetch(A + (size_t)(bm + (tid >> 1)) * K + k0 + 32, 0, 1);
#endif
    __syncthreads();

    // ---- fragments per ISA 7.12.2 layouts ----
    FragB a[2], b[2];
    const int kpA = hi * 4;  // A: lanes16-31 start at K elem 8 -> kpair 4
    const int kpB = hi * 4;  // B: lanes16-31 start at K elem 8 -> kpair 4
#pragma unroll
    for (int i = 0; i < 2; ++i) {
      const u32* p = &As32[cur][(warpM + i * 16 + l15) * 17 + kpA];
#pragma unroll
      for (int j = 0; j < 4; ++j) { a[i].u[j] = p[j]; a[i].u[4 + j] = p[8 + j]; }
    }
#pragma unroll
    for (int i = 0; i < 2; ++i) {
      const u32* p = &Bs32[(warpN + i * 16 + l15) * 17 + kpB];
#pragma unroll
      for (int j = 0; j < 4; ++j) { b[i].u[j] = p[j]; b[i].u[4 + j] = p[8 + j]; }
    }
#pragma unroll
    for (int i = 0; i < 2; ++i)
#pragma unroll
      for (int j = 0; j < 2; ++j)
        acc[i][j] = __builtin_amdgcn_wmma_f32_16x16x32_bf16(
            false, a[i].v, false, b[j].v, (short)0, acc[i][j], false, false);
  }

  // ---- epilogue (C layout: VGPR r, lane l -> m = r + 8*(l>=16), n = l&15) ----
  const int mloc = hi * 8;
#pragma unroll
  for (int i = 0; i < 2; ++i) {
#pragma unroll
    for (int j = 0; j < 2; ++j) {
#pragma unroll
      for (int r = 0; r < 8; ++r) {
        int gm = bm + warpM + i * 16 + mloc + r;
        int gn = bn + warpN + j * 16 + l15;
        float v = acc[i][j][r];
        if (MODE == 0) {
          outF[(size_t)gm * N + gn] = v + bias[gn];
        } else if (MODE == 1) {
          outB[(size_t)gm * N + gn] = f2bf(v);
        } else if (MODE == 2) {
          float g = mod[(size_t)(gm >> 5) * 3072 + gate_off + gn];
          outF[(size_t)gm * N + gn] += g * (v + bias[gn]);
        } else {
          float z  = v + bias[gn];
          float ge = 0.5f * z * (1.0f + erff(z * 0.70710678118f));
          outB[(size_t)gm * N + gn] = f2bf(ge);
        }
      }
    }
  }
}

// ---------------------------------------------------------------------------
// Flash attention, block-causal + key padding mask, diagonal always allowed.
// Grid (L/64, H, B), 128 threads = 4 waves; wave w owns 16 query rows.
// qkv layout: (B*L, 1536) bf16; q at +0, k at +512, v at +1024, head h at h*64.
// out layout: (B*L, 512) bf16 with col = h*64 + dh.
// ---------------------------------------------------------------------------
__global__ __launch_bounds__(128) void attn_kernel(
    const u16b* __restrict__ qkv, const unsigned char* __restrict__ mask,
    u16b* __restrict__ out) {
  __shared__ u32 Vs32[64 * 33];      // V^T: [dh][key-pair], stride 33
  __shared__ u32 Ps32[4 * 16 * 33];  // P per wave: [row][key-pair]

  const int lane  = threadIdx.x & 31;
  const int w     = threadIdx.x >> 5;
  const int hi    = lane >> 4;
  const int l15   = lane & 15;
  const int qt    = blockIdx.x;
  const int h     = blockIdx.y;
  const int b     = blockIdx.z;
  const int qbase = qt * 64;
  const size_t rowbase = (size_t)b * 1024;

  // ---- Q fragments straight from global (dh pairs are contiguous) ----
  FragB aQ[2];
  {
    int qi = qbase + w * 16 + l15;
    const u16b* qp = qkv + (rowbase + qi) * 1536 + h * 64;
#pragma unroll
    for (int s = 0; s < 2; ++s) {
      int dhb = s * 32 + hi * 8;
      uint4 d0 = *(const uint4*)(qp + dhb);
      uint4 d1 = *(const uint4*)(qp + dhb + 16);
      aQ[s].u[0] = d0.x; aQ[s].u[1] = d0.y; aQ[s].u[2] = d0.z; aQ[s].u[3] = d0.w;
      aQ[s].u[4] = d1.x; aQ[s].u[5] = d1.y; aQ[s].u[6] = d1.z; aQ[s].u[7] = d1.w;
    }
  }

  float m_i[8], l_i[8];
  v8f o[4];
#pragma unroll
  for (int r = 0; r < 8; ++r) { m_i[r] = -3.0e38f; l_i[r] = 0.f; }
#pragma unroll
  for (int d = 0; d < 4; ++d) o[d] = zero_v8f();

  const float scale = 0.125f;  // 1/sqrt(DH=64)
  const int qiBase = qbase + w * 16 + hi * 8;

  for (int kt = 0; kt <= qt; ++kt) {  // block-causal: later key tiles skipped
    const int kbase = kt * 64;
    __syncthreads();  // protect Vs from previous iteration's readers
    // ---- stage V tile transposed: Vs[dh][key] ----
    for (int c = threadIdx.x; c < 512; c += 128) {
      int key = c >> 3;
      int d8  = (c & 7) * 8;
      uint4 d = *(const uint4*)(qkv + (rowbase + kbase + key) * 1536 + 1024 + h * 64 + d8);
      u16b tmp[8];
      *(uint4*)tmp = d;
      u16b* Vsh = (u16b*)Vs32;
#pragma unroll
      for (int j = 0; j < 8; ++j) Vsh[(d8 + j) * 66 + key] = tmp[j];
    }
    __syncthreads();

    // ---- scores S = Q K^T (K fragments straight from global) ----
    v8f sc[4];
#pragma unroll
    for (int sub = 0; sub < 4; ++sub) {
      v8f s = zero_v8f();
      int key = kbase + sub * 16 + l15;
      const u16b* kp = qkv + (rowbase + key) * 1536 + 512 + h * 64;
#pragma unroll
      for (int s0 = 0; s0 < 2; ++s0) {
        FragB bK;
        int dhb = s0 * 32 + hi * 8;
        uint4 d0 = *(const uint4*)(kp + dhb);
        uint4 d1 = *(const uint4*)(kp + dhb + 16);
        bK.u[0] = d0.x; bK.u[1] = d0.y; bK.u[2] = d0.z; bK.u[3] = d0.w;
        bK.u[4] = d1.x; bK.u[5] = d1.y; bK.u[6] = d1.z; bK.u[7] = d1.w;
        s = __builtin_amdgcn_wmma_f32_16x16x32_bf16(
            false, aQ[s0].v, false, bK.v, (short)0, s, false, false);
      }
      sc[sub] = s;
    }

    // ---- mask + scale ----
#pragma unroll
    for (int sub = 0; sub < 4; ++sub) {
      int kj = kbase + sub * 16 + l15;
      int tk = kj >> 5;
      bool pad = mask[((size_t)b * 32 + tk) * 32 + (kj & 31)] != 0;
#pragma unroll
      for (int r = 0; r < 8; ++r) {
        int qi = qiBase + r;
        int tq = qi >> 5;
        bool ok = ((tq >= tk) && pad) || (qi == kj);
        float v = sc[sub][r] * scale;
        sc[sub][r] = ok ? v : -1.0e30f;
      }
    }

    // ---- online softmax update (row reductions over 16-lane halves) ----
#pragma unroll
    for (int r = 0; r < 8; ++r) {
      float v = fmaxf(fmaxf(sc[0][r], sc[1][r]), fmaxf(sc[2][r], sc[3][r]));
#pragma unroll
      for (int off = 8; off > 0; off >>= 1) v = fmaxf(v, __shfl_xor(v, off, 32));
      float mnew  = fmaxf(m_i[r], v);
      float alpha = __expf(m_i[r] - mnew);
      m_i[r] = mnew;
      float rs = 0.f;
#pragma unroll
      for (int sub = 0; sub < 4; ++sub) {
        float p = __expf(sc[sub][r] - mnew);
        sc[sub][r] = p;
        rs += p;
      }
#pragma unroll
      for (int off = 8; off > 0; off >>= 1) rs += __shfl_xor(rs, off, 32);
      l_i[r] = l_i[r] * alpha + rs;
#pragma unroll
      for (int d = 0; d < 4; ++d) o[d][r] *= alpha;
    }

    // ---- transpose P through LDS (C layout -> A layout) ----
    {
      u16b* Psh = (u16b*)Ps32;
      int mbase = w * 16 + hi * 8;
#pragma unroll
      for (int sub = 0; sub < 4; ++sub)
#pragma unroll
        for (int r = 0; r < 8; ++r)
          Psh[(mbase + r) * 66 + sub * 16 + l15] = f2bf(sc[sub][r]);
    }
    FragB aP[2];
#pragma unroll
    for (int s0 = 0; s0 < 2; ++s0) {
      const u32* p = &Ps32[(w * 16 + l15) * 33 + s0 * 16 + hi * 4];
#pragma unroll
      for (int j = 0; j < 4; ++j) { aP[s0].u[j] = p[j]; aP[s0].u[4 + j] = p[8 + j]; }
    }

    // ---- O += P @ V ----
#pragma unroll
    for (int d = 0; d < 4; ++d) {
#pragma unroll
      for (int s0 = 0; s0 < 2; ++s0) {
        FragB bV;
        const u32* p = &Vs32[(d * 16 + l15) * 33 + s0 * 16 + hi * 4];
#pragma unroll
        for (int j = 0; j < 4; ++j) { bV.u[j] = p[j]; bV.u[4 + j] = p[8 + j]; }
        o[d] = __builtin_amdgcn_wmma_f32_16x16x32_bf16(
            false, aP[s0].v, false, bV.v, (short)0, o[d], false, false);
      }
    }
  }

  // ---- normalize and store bf16 ----
#pragma unroll
  for (int d = 0; d < 4; ++d)
#pragma unroll
    for (int r = 0; r < 8; ++r) {
      float v = o[d][r] / l_i[r];
      out[(rowbase + qiBase + r) * 512 + h * 64 + d * 16 + l15] = f2bf(v);
    }
}

// ---------------------------------------------------------------------------
// LayerNorm helpers (one row of 512 per 256-thread block)
// ---------------------------------------------------------------------------
__device__ __forceinline__ float block_sum256(float v, float* red) {
#pragma unroll
  for (int o = 16; o > 0; o >>= 1) v += __shfl_xor(v, o, 32);
  int wv = threadIdx.x >> 5;
  __syncthreads();
  if ((threadIdx.x & 31) == 0) red[wv] = v;
  __syncthreads();
  float t = 0.f;
#pragma unroll
  for (int j = 0; j < 8; ++j) t += red[j];
  return t;
}

// h = ln(x)*(1+sc)+sh ; if SECOND: h = ln2(h)*g2+b2 ; out bf16
template <bool SECOND>
__global__ __launch_bounds__(256) void ln_mod_kernel(
    const float* __restrict__ x, const float* __restrict__ mod,
    const float* __restrict__ g1, const float* __restrict__ b1,
    const float* __restrict__ g2, const float* __restrict__ b2,
    int sh_off, int sc_off, u16b* __restrict__ out) {
  __shared__ float red[8];
  const int row = blockIdx.x;
  const int bt  = row >> 5;  // N = 32 objects per frame
  const int c0 = threadIdx.x, c1 = threadIdx.x + 256;
  const float* xr = x + (size_t)row * 512;
  float v0 = xr[c0], v1 = xr[c1];
  float mu  = block_sum256(v0 + v1, red) * (1.f / 512.f);
  float d0 = v0 - mu, d1 = v1 - mu;
  float var = block_sum256(d0 * d0 + d1 * d1, red) * (1.f / 512.f);
  float rs  = rsqrtf(var + 1e-5f);
  const float* mrow = mod + (size_t)bt * 3072;
  float h0 = (d0 * rs * g1[c0] + b1[c0]) * (1.f + mrow[sc_off + c0]) + mrow[sh_off + c0];
  float h1 = (d1 * rs * g1[c1] + b1[c1]) * (1.f + mrow[sc_off + c1]) + mrow[sh_off + c1];
  if (SECOND) {
    float mu2  = block_sum256(h0 + h1, red) * (1.f / 512.f);
    float e0 = h0 - mu2, e1 = h1 - mu2;
    float var2 = block_sum256(e0 * e0 + e1 * e1, red) * (1.f / 512.f);
    float rs2  = rsqrtf(var2 + 1e-5f);
    h0 = e0 * rs2 * g2[c0] + b2[c0];
    h1 = e1 * rs2 * g2[c1] + b2[c1];
  }
  u16b* orow = out + (size_t)row * 512;
  orow[c0] = f2bf(h0);
  orow[c1] = f2bf(h1);
}

__global__ __launch_bounds__(256) void final_ln_kernel(
    const float* __restrict__ x, const float* __restrict__ g,
    const float* __restrict__ bb, float* __restrict__ out) {
  __shared__ float red[8];
  const int row = blockIdx.x;
  const int c0 = threadIdx.x, c1 = threadIdx.x + 256;
  const float* xr = x + (size_t)row * 512;
  float v0 = xr[c0], v1 = xr[c1];
  float mu  = block_sum256(v0 + v1, red) * (1.f / 512.f);
  float d0 = v0 - mu, d1 = v1 - mu;
  float var = block_sum256(d0 * d0 + d1 * d1, red) * (1.f / 512.f);
  float rs  = rsqrtf(var + 1e-5f);
  out[(size_t)row * 512 + c0] = d0 * rs * g[c0] + bb[c0];
  out[(size_t)row * 512 + c1] = d1 * rs * g[c1] + bb[c1];
}

// ---------------------------------------------------------------------------
// Elementwise kernels
// ---------------------------------------------------------------------------
__global__ void prep_kernel(const float* __restrict__ x, const float* __restrict__ tpos,
                            const float* __restrict__ opos, float* __restrict__ xo,
                            int total) {
  int i = blockIdx.x * 256 + threadIdx.x;
  if (i >= total) return;
  int d = i & 511;
  int tn = i >> 9;
  int n = tn & 31;
  int t = (tn >> 5) & 31;
  xo[i] = x[i] + tpos[t * 512 + d] + opos[n * 512 + d];
}

__global__ void silu_bf16_kernel(const float* __restrict__ in, u16b* __restrict__ out, int n) {
  int i = blockIdx.x * 256 + threadIdx.x;
  if (i >= n) return;
  float z = in[i];
  out[i] = f2bf(z / (1.f + __expf(-z)));
}

__global__ void f32_to_bf16_kernel(const float* __restrict__ in, u16b* __restrict__ out, int n) {
  int i = blockIdx.x * 256 + threadIdx.x;
  if (i < n) out[i] = f2bf(in[i]);
}

// ---------------------------------------------------------------------------
// Host orchestration
// ---------------------------------------------------------------------------
extern "C" void kernel_launch(void* const* d_in, const int* in_sizes, int n_in,
                              void* d_out, int out_size, void* d_ws, size_t ws_size,
                              hipStream_t stream) {
  (void)in_sizes; (void)n_in; (void)out_size; (void)ws_size;

  const float* x_in   = (const float*)d_in[0];
  const float* c_in   = (const float*)d_in[1];
  const unsigned char* mask = (const unsigned char*)d_in[2];  // jax bool = 1 byte
  const float* tpos   = (const float*)d_in[3];
  const float* opos   = (const float*)d_in[4];
  const float* ln1_g  = (const float*)d_in[5];
  const float* ln1_b  = (const float*)d_in[6];
  const float* ln2_g  = (const float*)d_in[7];
  const float* ln2_b  = (const float*)d_in[8];
  const float* aln_g  = (const float*)d_in[9];
  const float* aln_b  = (const float*)d_in[10];
  const float* ada_w  = (const float*)d_in[11];
  const float* ada_b  = (const float*)d_in[12];
  const float* w_qkv  = (const float*)d_in[13];
  const float* w_out  = (const float*)d_in[14];
  const float* b_out  = (const float*)d_in[15];
  const float* mlp_w1 = (const float*)d_in[16];
  const float* mlp_b1 = (const float*)d_in[17];
  const float* mlp_w2 = (const float*)d_in[18];
  const float* mlp_b2 = (const float*)d_in[19];
  const float* fin_g  = (const float*)d_in[20];
  const float* fin_b  = (const float*)d_in[21];

  char* ws = (char*)d_ws;
  size_t off = 0;
  auto take = [&](size_t bytes) -> char* {
    char* p = ws + off;
    off += (bytes + 255) & ~(size_t)255;
    return p;
  };
  float* xbuf   = (float*)take((size_t)8192 * 512 * 4);   // residual stream (f32)
  float* modbuf = (float*)take((size_t)256 * 3072 * 4);   // adaLN modulation
  u16b*  cfb    = (u16b*)take((size_t)256 * 512 * 2);     // silu(c) bf16
  u16b*  hnbuf  = (u16b*)take((size_t)8192 * 512 * 2);    // LN out; aliased w/ attn out
  u16b*  bigbuf = (u16b*)take((size_t)8192 * 2048 * 2);   // qkv (x1536) then mlp act (x2048)
  u16b*  wada   = (u16b*)take((size_t)4 * 512 * 3072 * 2);
  u16b*  wqkvb  = (u16b*)take((size_t)4 * 512 * 1536 * 2);
  u16b*  woutb  = (u16b*)take((size_t)4 * 512 * 512 * 2);
  u16b*  wm1b   = (u16b*)take((size_t)4 * 512 * 2048 * 2);
  u16b*  wm2b   = (u16b*)take((size_t)4 * 2048 * 512 * 2);

  auto cvt = [&](const float* src, u16b* dst, int n) {
    f32_to_bf16_kernel<<<(n + 255) / 256, 256, 0, stream>>>(src, dst, n);
  };
  cvt(ada_w,  wada,  4 * 512 * 3072);
  cvt(w_qkv,  wqkvb, 4 * 512 * 1536);
  cvt(w_out,  woutb, 4 * 512 * 512);
  cvt(mlp_w1, wm1b,  4 * 512 * 2048);
  cvt(mlp_w2, wm2b,  4 * 2048 * 512);

  const int total = 8 * 1024 * 512;
  prep_kernel<<<total / 256, 256, 0, stream>>>(x_in, tpos, opos, xbuf, total);
  silu_bf16_kernel<<<(256 * 512) / 256, 256, 0, stream>>>(c_in, cfb, 256 * 512);

  for (int i = 0; i < 4; ++i) {
    // mod = silu(c) @ ada_w + ada_b          (256 x 3072, K=512)
    gemm_bf16<0><<<dim3(4, 48), 128, 0, stream>>>(
        cfb, wada + (size_t)i * 512 * 3072, 256, 3072, 512,
        ada_b + (size_t)i * 3072, nullptr, 0, modbuf, nullptr);
    // hn = attn_ln(modulate(ln1(x)))
    ln_mod_kernel<true><<<8192, 256, 0, stream>>>(
        xbuf, modbuf, ln1_g + i * 512, ln1_b + i * 512,
        aln_g + i * 512, aln_b + i * 512, /*sh*/0, /*sc*/512, hnbuf);
    // qkv = hn @ w_qkv                      (8192 x 1536, K=512)
    gemm_bf16<1><<<dim3(128, 24), 128, 0, stream>>>(
        hnbuf, wqkvb + (size_t)i * 512 * 1536, 8192, 1536, 512,
        nullptr, nullptr, 0, nullptr, bigbuf);
    // attention (flash, block-causal + padding mask), writes into hnbuf
    attn_kernel<<<dim3(16, 8, 8), 128, 0, stream>>>(bigbuf, mask, hnbuf);
    // x += g_msa * (o @ w_out + b_out)       (8192 x 512, K=512), gate chunk 2
    gemm_bf16<2><<<dim3(128, 8), 128, 0, stream>>>(
        hnbuf, woutb + (size_t)i * 512 * 512, 8192, 512, 512,
        b_out + i * 512, modbuf, 2 * 512, xbuf, nullptr);
    // h2 = modulate(ln2(x))
    ln_mod_kernel<false><<<8192, 256, 0, stream>>>(
        xbuf, modbuf, ln2_g + i * 512, ln2_b + i * 512,
        nullptr, nullptr, /*sh*/3 * 512, /*sc*/4 * 512, hnbuf);
    // act = gelu(h2 @ mlp_w1 + b1)           (8192 x 2048, K=512)
    gemm_bf16<3><<<dim3(128, 32), 128, 0, stream>>>(
        hnbuf, wm1b + (size_t)i * 512 * 2048, 8192, 2048, 512,
        mlp_b1 + (size_t)i * 2048, nullptr, 0, nullptr, bigbuf);
    // x += g_mlp * (act @ mlp_w2 + b2)       (8192 x 512, K=2048), gate chunk 5
    gemm_bf16<2><<<dim3(128, 8), 128, 0, stream>>>(
        bigbuf, wm2b + (size_t)i * 2048 * 512, 8192, 512, 2048,
        mlp_b2 + i * 512, modbuf, 5 * 512, xbuf, nullptr);
  }

  final_ln_kernel<<<8192, 256, 0, stream>>>(xbuf, fin_g, fin_b, (float*)d_out);
}
